// MSA_yolov_online_4664334483747
// MI455X (gfx1250) — matrix operand order repaired and checked by reading, not compiled, not run on hardware
//
#include <hip/hip_runtime.h>
#include <hip/hip_bf16.h>

typedef __bf16 bf16_t;
typedef __attribute__((ext_vector_type(16))) __bf16 v16bf;
typedef __attribute__((ext_vector_type(8)))  __bf16 v8bf;
typedef __attribute__((ext_vector_type(8)))  float  v8f;

constexpr int Nn  = 2048;   // tokens
constexpr int Cc  = 1024;   // channels
constexpr int Hh  = 8;      // heads
constexpr int Dd  = 128;    // head dim
constexpr int C2  = 2048;
constexpr int C3  = 3072;
constexpr int C4  = 4096;
constexpr int OUTC = 1024;
constexpr float SCALE = 25.0f;
constexpr float MASK_THR_xH = 0.75f * 8.0f;   // threshold vs un-divided head sum

// ---------------------------------------------------------------------------
// CDNA5 async global->LDS copy (ASYNCcnt-tracked) + wait, via inline asm.
// lds_off is a byte offset from the workgroup LDS base (sole __shared__ array).
// ---------------------------------------------------------------------------
__device__ __forceinline__ void async_g2l_b128(unsigned lds_off, const bf16_t* g)
{
    asm volatile("global_load_async_to_lds_b128 %0, %1, off"
                 :: "v"(lds_off), "v"(g) : "memory");
}
__device__ __forceinline__ void wait_async0()
{
    asm volatile("s_wait_asynccnt 0x0" ::: "memory");
}

// ---------------------------------------------------------------------------
// BF16 WMMA GEMM:  C[M,Ncol] = A[M,K](row-major) * B[Ncol,K](row-major)^T
// optional bias[Ncol], optional accumulate into existing C.
// Block = 256 threads (8 waves), block tile 128x128, wave tile 32x64.
// K staged through LDS in 32-wide double-buffered slabs via async copies.
// ---------------------------------------------------------------------------
constexpr int BK      = 32;            // K slab
constexpr int LSTRIDE = BK + 8;        // padded LDS row stride (elems) -> 20 banks
constexpr int TILE_E  = 128 * LSTRIDE; // elems per A or B tile buffer

__global__ __launch_bounds__(256)
void gemm_bf16_wmma(const bf16_t* __restrict__ A, int lda,
                    const bf16_t* __restrict__ B, int ldb,
                    float* __restrict__ C, int ldc,
                    int K, const float* __restrict__ bias, int accumulate)
{
    __shared__ bf16_t lds[4 * TILE_E];     // [buf0:A,B][buf1:A,B]

    const int t = threadIdx.x;
    const unsigned lane = t & 31u;
    const unsigned wave = t >> 5;
    const int wm = (int)(wave >> 1);       // 0..3 (M)
    const int wn = (int)(wave & 1u);       // 0..1 (N)
    const int mblk = blockIdx.y * 128;
    const int nblk = blockIdx.x * 128;

    const int half = (int)(lane >> 4);     // 0: lanes 0-15, 1: lanes 16-31
    const int lrow = (int)(lane & 15u);

    // this thread's staging coords: 128 rows x 32 elems(64B) = 512 x 16B chunks
    const int c0   = t;                    // chunk ids t and t+256
    // staging lambda: issue 4 async b128 per thread (2 for A, 2 for B)
    auto stage = [&](int buf, int k0) {
        const bf16_t* gA = A + (size_t)mblk * lda + k0;
        const bf16_t* gB = B + (size_t)nblk * ldb + k0;
        const unsigned base = (unsigned)(buf * 2 * TILE_E) * 2u;   // bytes
        #pragma unroll
        for (int q = 0; q < 2; ++q) {
            const int c   = c0 + q * 256;
            const int row = c >> 2;            // 4 chunks per row
            const int col = (c & 3) * 8;       // elems
            const unsigned lo = (unsigned)(row * LSTRIDE + col) * 2u;
            async_g2l_b128(base + lo,                     gA + (size_t)row * lda + col);
            async_g2l_b128(base + (unsigned)TILE_E * 2u + lo, gB + (size_t)row * ldb + col);
        }
    };

    v8f acc[2][4];
    const int crow0 = half * 8;
    if (accumulate) {
        #pragma unroll
        for (int i = 0; i < 2; ++i)
            #pragma unroll
            for (int j = 0; j < 4; ++j) {
                const int col = nblk + wn * 64 + 16 * j + lrow;
                #pragma unroll
                for (int e = 0; e < 8; ++e)
                    acc[i][j][e] = C[(size_t)(mblk + wm * 32 + 16 * i + crow0 + e) * ldc + col];
            }
    } else {
        #pragma unroll
        for (int i = 0; i < 2; ++i)
            #pragma unroll
            for (int j = 0; j < 4; ++j)
                #pragma unroll
                for (int e = 0; e < 8; ++e)
                    acc[i][j][e] = 0.0f;
    }

    const int nk = K / BK;
    stage(0, 0);
    for (int ki = 0; ki < nk; ++ki) {
        const int buf = ki & 1;
        wait_async0();           // own async copies landed in LDS
        __syncthreads();         // everyone's copies visible
        if (ki + 1 < nk) stage(buf ^ 1, (ki + 1) * BK);

        const bf16_t* tA = lds + buf * 2 * TILE_E;
        const bf16_t* tB = tA + TILE_E;
        // A fragments (16x32): lane<16 holds K {0..7,16..23}, lane>=16 +8
        v16bf a[2];
        #pragma unroll
        for (int i = 0; i < 2; ++i) {
            const bf16_t* p = tA + (wm * 32 + 16 * i + lrow) * LSTRIDE + half * 8;
            v8bf vlo = *(const v8bf*)(p);
            v8bf vhi = *(const v8bf*)(p + 16);
            #pragma unroll
            for (int e = 0; e < 8; ++e) { a[i][e] = vlo[e]; a[i][8 + e] = vhi[e]; }
        }
        // B fragments (32x16): lane = column, lane<16 holds K {0..7,16..23}, lane>=16 +8
        v16bf b[4];
        #pragma unroll
        for (int j = 0; j < 4; ++j) {
            const bf16_t* p = tB + (wn * 64 + 16 * j + lrow) * LSTRIDE + half * 8;
            v8bf vlo = *(const v8bf*)(p);
            v8bf vhi = *(const v8bf*)(p + 16);
            #pragma unroll
            for (int e = 0; e < 8; ++e) { b[j][e] = vlo[e]; b[j][8 + e] = vhi[e]; }
        }
        #pragma unroll
        for (int i = 0; i < 2; ++i)
            #pragma unroll
            for (int j = 0; j < 4; ++j)
                acc[i][j] = __builtin_amdgcn_wmma_f32_16x16x32_bf16(
                    false, a[i], false, b[j], (short)0, acc[i][j], false, false);
        __syncthreads();         // protect buffer reuse
    }

    #pragma unroll
    for (int i = 0; i < 2; ++i)
        #pragma unroll
        for (int j = 0; j < 4; ++j) {
            const int col = nblk + wn * 64 + 16 * j + lrow;
            const float bv = bias ? bias[col] : 0.0f;
            #pragma unroll
            for (int e = 0; e < 8; ++e)
                C[(size_t)(mblk + wm * 32 + 16 * i + crow0 + e) * ldc + col] = acc[i][j][e] + bv;
        }
}

// ---------------------------------------------------------------------------
__global__ void cvt_bf16_kernel(const float* __restrict__ src, bf16_t* __restrict__ dst, int n)
{
    int i = blockIdx.x * blockDim.x + threadIdx.x;
    if (i < n) dst[i] = (bf16_t)src[i];
}

__global__ void zero_f32_kernel(float* __restrict__ dst, int n)
{
    int i = blockIdx.x * blockDim.x + threadIdx.x;
    if (i < n) dst[i] = 0.0f;
}

// ---------------------------------------------------------------------------
// qkv [N,3C] f32 -> per-head L2-normalized bf16 operand layouts
// ---------------------------------------------------------------------------
__global__ __launch_bounds__(128)
void reorg_norm_cls_kernel(const float* __restrict__ qkv,
                           bf16_t* __restrict__ qn, bf16_t* __restrict__ kn,
                           bf16_t* __restrict__ vn,
                           bf16_t* __restrict__ vt, float* __restrict__ xori)
{
    const int n = blockIdx.x, h = blockIdx.y, dd = threadIdx.x;
    const size_t base = (size_t)n * C3 + (size_t)h * Dd + dd;
    const float q = qkv[base];
    const float k = qkv[base + Cc];
    const float v = qkv[base + 2 * Cc];
    __shared__ float sq[128], sk[128], sv[128];
    sq[dd] = q * q; sk[dd] = k * k; sv[dd] = v * v;
    __syncthreads();
    for (int st = 64; st > 0; st >>= 1) {
        if (dd < st) { sq[dd] += sq[dd + st]; sk[dd] += sk[dd + st]; sv[dd] += sv[dd + st]; }
        __syncthreads();
    }
    const float rq = rsqrtf(sq[0]), rk = rsqrtf(sk[0]), rv = rsqrtf(sv[0]);
    const size_t o = ((size_t)h * Nn + n) * Dd + dd;
    qn[o] = (bf16_t)(q * rq);
    kn[o] = (bf16_t)(k * rk);
    vn[o] = (bf16_t)(v * rv);
    vt[((size_t)h * Dd + dd) * Nn + n] = (bf16_t)v;           // v^T for attn@v B-operand
    xori[(size_t)n * Cc + (size_t)h * Dd + dd] = v;           // x_ori
}

__global__ __launch_bounds__(128)
void reorg_norm_reg_kernel(const float* __restrict__ qkv,
                           bf16_t* __restrict__ qn, bf16_t* __restrict__ kn)
{
    const int n = blockIdx.x, h = blockIdx.y, dd = threadIdx.x;
    const size_t base = (size_t)n * C3 + (size_t)h * Dd + dd;
    const float q = qkv[base];
    const float k = qkv[base + Cc];
    __shared__ float sq[128], sk[128];
    sq[dd] = q * q; sk[dd] = k * k;
    __syncthreads();
    for (int st = 64; st > 0; st >>= 1) {
        if (dd < st) { sq[dd] += sq[dd + st]; sk[dd] += sk[dd + st]; }
        __syncthreads();
    }
    const float rq = rsqrtf(sq[0]), rk = rsqrtf(sk[0]);
    const size_t o = ((size_t)h * Nn + n) * Dd + dd;
    qn[o] = (bf16_t)(q * rq);
    kn[o] = (bf16_t)(k * rk);
}

// ---------------------------------------------------------------------------
// Per-head dual softmax: attn = 0.5*(softmax(Lc*25*cs) + softmax(Lr*25*fs))
// ---------------------------------------------------------------------------
__global__ __launch_bounds__(256)
void attn_softmax_kernel(const float* __restrict__ Lc, const float* __restrict__ Lr,
                         const float* __restrict__ cls_score, const float* __restrict__ fg_score,
                         bf16_t* __restrict__ attn_h, float* __restrict__ attn_sum)
{
    const int n = blockIdx.x, t = threadIdx.x;
    float vc[8], vr[8];
    float mc = -1e30f, mr = -1e30f;
    #pragma unroll
    for (int i = 0; i < 8; ++i) {
        const int m = t + i * 256;
        const float c = Lc[(size_t)n * Nn + m] * (SCALE * cls_score[m]);
        const float r = Lr[(size_t)n * Nn + m] * (SCALE * fg_score[m]);
        vc[i] = c; vr[i] = r;
        mc = fmaxf(mc, c); mr = fmaxf(mr, r);
    }
    __shared__ float s0[256], s1[256];
    s0[t] = mc; s1[t] = mr; __syncthreads();
    for (int st = 128; st > 0; st >>= 1) {
        if (t < st) { s0[t] = fmaxf(s0[t], s0[t + st]); s1[t] = fmaxf(s1[t], s1[t + st]); }
        __syncthreads();
    }
    mc = s0[0]; mr = s1[0]; __syncthreads();
    float sc = 0.0f, sr = 0.0f;
    #pragma unroll
    for (int i = 0; i < 8; ++i) {
        vc[i] = __expf(vc[i] - mc); sc += vc[i];
        vr[i] = __expf(vr[i] - mr); sr += vr[i];
    }
    s0[t] = sc; s1[t] = sr; __syncthreads();
    for (int st = 128; st > 0; st >>= 1) {
        if (t < st) { s0[t] += s0[t + st]; s1[t] += s1[t + st]; }
        __syncthreads();
    }
    const float ic = 1.0f / s0[0], ir = 1.0f / s1[0];
    #pragma unroll
    for (int i = 0; i < 8; ++i) {
        const int m = t + i * 256;
        const float pv = 0.5f * (vc[i] * ic + vr[i] * ir);
        attn_h[(size_t)n * Nn + m] = (bf16_t)pv;
        attn_sum[(size_t)n * Nn + m] += pv;
    }
}

// ---------------------------------------------------------------------------
// sim_round2 = softmax(attn_sum/H); masked by (sim_raw_sum > 0.75*H); renorm.
// ---------------------------------------------------------------------------
__global__ __launch_bounds__(256)
void sim_mask_softmax_kernel(const float* __restrict__ attn_sum,
                             const float* __restrict__ simacc,
                             bf16_t* __restrict__ simw)
{
    const int n = blockIdx.x, t = threadIdx.x;
    float a[8];
    float mx = -1e30f;
    #pragma unroll
    for (int i = 0; i < 8; ++i) {
        a[i] = attn_sum[(size_t)n * Nn + t + i * 256] * (1.0f / Hh);
        mx = fmaxf(mx, a[i]);
    }
    __shared__ float s0[256];
    s0[t] = mx; __syncthreads();
    for (int st = 128; st > 0; st >>= 1) {
        if (t < st) s0[t] = fmaxf(s0[t], s0[t + st]);
        __syncthreads();
    }
    mx = s0[0]; __syncthreads();
    float sum = 0.0f;
    #pragma unroll
    for (int i = 0; i < 8; ++i) { a[i] = __expf(a[i] - mx); sum += a[i]; }
    s0[t] = sum; __syncthreads();
    for (int st = 128; st > 0; st >>= 1) {
        if (t < st) s0[t] += s0[t + st];
        __syncthreads();
    }
    const float inv = 1.0f / s0[0]; __syncthreads();
    float pm[8];
    float msum = 0.0f;
    #pragma unroll
    for (int i = 0; i < 8; ++i) {
        const int m = t + i * 256;
        const float pv = a[i] * inv;
        pm[i] = (simacc[(size_t)n * Nn + m] > MASK_THR_xH) ? pv : 0.0f;
        msum += pm[i];
    }
    s0[t] = msum; __syncthreads();
    for (int st = 128; st > 0; st >>= 1) {
        if (t < st) s0[t] += s0[t + st];
        __syncthreads();
    }
    const float r = 1.0f / s0[0];
    #pragma unroll
    for (int i = 0; i < 8; ++i)
        simw[(size_t)n * Nn + t + i * 256] = (bf16_t)(pm[i] * r);
}

// ---------------------------------------------------------------------------
// transpose + convert: dst[j][m] = (bf16) src[m][j]
// ---------------------------------------------------------------------------
__global__ __launch_bounds__(256)
void transpose_cvt_kernel(const float* __restrict__ src, bf16_t* __restrict__ dst,
                          int rows, int cols)
{
    __shared__ float tile[32][33];
    const int bx = blockIdx.x * 32;
    const int by = blockIdx.y * 32;
    const int tx = threadIdx.x & 31, ty = threadIdx.x >> 5;
    #pragma unroll
    for (int r = 0; r < 4; ++r)
        tile[ty + r * 8][tx] = src[(size_t)(by + ty + r * 8) * cols + bx + tx];
    __syncthreads();
    #pragma unroll
    for (int r = 0; r < 4; ++r)
        dst[(size_t)(bx + ty + r * 8) * rows + by + tx] = (bf16_t)tile[tx][ty + r * 8];
}

// ---------------------------------------------------------------------------
__global__ void concat2_cvt_kernel(const float* __restrict__ left, const float* __restrict__ right,
                                   bf16_t* __restrict__ out, int halfcols, int total)
{
    int i = blockIdx.x * blockDim.x + threadIdx.x;
    if (i >= total) return;
    const int w = 2 * halfcols;
    const int r = i / w, c = i % w;
    const float v = (c < halfcols) ? left[(size_t)r * halfcols + c]
                                   : right[(size_t)r * halfcols + (c - halfcols)];
    out[i] = (bf16_t)v;
}

// ---------------------------------------------------------------------------
extern "C" void kernel_launch(void* const* d_in, const int* in_sizes, int n_in,
                              void* d_out, int out_size, void* d_ws, size_t ws_size,
                              hipStream_t stream)
{
    const float* x_cls     = (const float*)d_in[0];
    const float* x_reg     = (const float*)d_in[1];
    const float* cls_score = (const float*)d_in[2];
    const float* fg_score  = (const float*)d_in[3];
    const float* Wqc       = (const float*)d_in[4];
    const float* Wqr       = (const float*)d_in[5];
    const float* W1        = (const float*)d_in[6];
    const float* b1        = (const float*)d_in[7];
    const float* W2        = (const float*)d_in[8];
    const float* b2        = (const float*)d_in[9];
    float* out = (float*)d_out;

    char* p = (char*)d_ws;
    auto carve = [&](size_t bytes) -> void* {
        void* r = (void*)p;
        p += (bytes + 255) & ~(size_t)255;
        return r;
    };
    bf16_t* xcls_bf = (bf16_t*)carve((size_t)Nn * Cc * 2);
    bf16_t* xreg_bf = (bf16_t*)carve((size_t)Nn * Cc * 2);
    bf16_t* wqc_bf  = (bf16_t*)carve((size_t)C3 * Cc * 2);
    bf16_t* wqr_bf  = (bf16_t*)carve((size_t)C3 * Cc * 2);
    bf16_t* w1_bf   = (bf16_t*)carve((size_t)C2 * C2 * 2);
    bf16_t* w2_bf   = (bf16_t*)carve((size_t)OUTC * C4 * 2);
    float*  qkvc    = (float*)carve((size_t)Nn * C3 * 4);
    float*  qkvr    = (float*)carve((size_t)Nn * C3 * 4);
    bf16_t* qcn     = (bf16_t*)carve((size_t)Hh * Nn * Dd * 2);
    bf16_t* kcn     = (bf16_t*)carve((size_t)Hh * Nn * Dd * 2);
    bf16_t* qrn     = (bf16_t*)carve((size_t)Hh * Nn * Dd * 2);
    bf16_t* krn     = (bf16_t*)carve((size_t)Hh * Nn * Dd * 2);
    bf16_t* vnb     = (bf16_t*)carve((size_t)Hh * Nn * Dd * 2);
    bf16_t* vt      = (bf16_t*)carve((size_t)Hh * Dd * Nn * 2);
    float*  xori    = (float*)carve((size_t)Nn * Cc * 4);
    float*  Lc      = (float*)carve((size_t)Nn * Nn * 4);
    float*  Lr      = (float*)carve((size_t)Nn * Nn * 4);
    bf16_t* attn_h  = (bf16_t*)carve((size_t)Nn * Nn * 2);
    float*  attn_sum= (float*)carve((size_t)Nn * Nn * 4);
    float*  simacc  = (float*)carve((size_t)Nn * Nn * 4);
    float*  xfull   = (float*)carve((size_t)Nn * Cc * 4);
    bf16_t* trans_bf= (bf16_t*)carve((size_t)Nn * C2 * 2);
    float*  msa     = (float*)carve((size_t)Nn * C2 * 4);
    bf16_t* featT   = (bf16_t*)carve((size_t)C2 * Nn * 2);
    bf16_t* simw    = (bf16_t*)carve((size_t)Nn * Nn * 2);
    float*  softsim = (float*)carve((size_t)Nn * C2 * 4);
    bf16_t* ave_bf  = (bf16_t*)carve((size_t)Nn * C4 * 2);

    auto cvt = [&](const float* s, bf16_t* d, int n) {
        cvt_bf16_kernel<<<(n + 255) / 256, 256, 0, stream>>>(s, d, n);
    };
    auto gemm = [&](const bf16_t* A, int lda, const bf16_t* B, int ldb,
                    float* Cp, int ldc, int M, int Ncol, int K,
                    const float* bias, int accum) {
        dim3 g(Ncol / 128, M / 128);
        gemm_bf16_wmma<<<g, 256, 0, stream>>>(A, lda, B, ldb, Cp, ldc, K, bias, accum);
    };

    // 1) bf16 copies of activations and weights
    cvt(x_cls, xcls_bf, Nn * Cc);
    cvt(x_reg, xreg_bf, Nn * Cc);
    cvt(Wqc,   wqc_bf,  C3 * Cc);
    cvt(Wqr,   wqr_bf,  C3 * Cc);
    cvt(W1,    w1_bf,   C2 * C2);
    cvt(W2,    w2_bf,   OUTC * C4);
    zero_f32_kernel<<<(Nn * Nn + 255) / 256, 256, 0, stream>>>(attn_sum, Nn * Nn);
    zero_f32_kernel<<<(Nn * Nn + 255) / 256, 256, 0, stream>>>(simacc,   Nn * Nn);

    // 2) QKV projections: qkv = x @ W^T   [2048 x 3072]
    gemm(xcls_bf, Cc, wqc_bf, Cc, qkvc, C3, Nn, C3, Cc, nullptr, 0);
    gemm(xreg_bf, Cc, wqr_bf, Cc, qkvr, C3, Nn, C3, Cc, nullptr, 0);

    // 3) split heads + L2 normalize
    reorg_norm_cls_kernel<<<dim3(Nn, Hh), 128, 0, stream>>>(qkvc, qcn, kcn, vnb, vt, xori);
    reorg_norm_reg_kernel<<<dim3(Nn, Hh), 128, 0, stream>>>(qkvr, qrn, krn);

    // 4) per-head attention
    for (int h = 0; h < Hh; ++h) {
        const size_t ho = (size_t)h * Nn * Dd;
        gemm(qcn + ho, Dd, kcn + ho, Dd, Lc, Nn, Nn, Nn, Dd, nullptr, 0);   // q_cls k_cls^T
        gemm(qrn + ho, Dd, krn + ho, Dd, Lr, Nn, Nn, Nn, Dd, nullptr, 0);   // q_reg k_reg^T
        attn_softmax_kernel<<<Nn, 256, 0, stream>>>(Lc, Lr, cls_score, fg_score, attn_h, attn_sum);
        // x_h = attn_h @ v  (B operand = v^T rows [d][N])
        gemm(attn_h, Nn, vt + (size_t)h * Dd * Nn, Nn, xfull + h * Dd, Cc, Nn, Dd, Nn, nullptr, 0);
        // sim_raw accumulation: simacc += vn @ vn^T
        gemm(vnb + ho, Dd, vnb + ho, Dd, simacc, Nn, Nn, Nn, Dd, nullptr, 1);
    }

    // 5) trans_cls = concat(x, x_ori) -> bf16; msa = trans_cls @ W1^T + b1
    concat2_cvt_kernel<<<(Nn * C2 + 255) / 256, 256, 0, stream>>>(xfull, xori, trans_bf, Cc, Nn * C2);
    gemm(trans_bf, C2, w1_bf, C2, msa, C2, Nn, C2, C2, b1, 0);

    // 6) masked renormalized similarity (bf16 for WMMA)
    sim_mask_softmax_kernel<<<Nn, 256, 0, stream>>>(attn_sum, simacc, simw);

    // 7) soft_sim = simw @ feat  (B operand = feat^T)
    transpose_cvt_kernel<<<dim3(C2 / 32, Nn / 32), 256, 0, stream>>>(msa, featT, Nn, C2);
    gemm(simw, Nn, featT, Nn, softsim, C2, Nn, C2, Nn, nullptr, 0);

    // 8) ave = concat(soft_sim, feat); out = ave @ W2^T + b2
    concat2_cvt_kernel<<<(Nn * C4 + 255) / 256, 256, 0, stream>>>(softsim, msa, ave_bf, C2, Nn * C4);
    gemm(ave_bf, C4, w2_bf, C4, out, OUTC, Nn, OUTC, C4, b2, 0);
}
